// EquivariantBlock_48275432407131
// MI455X (gfx1250) — matrix operand order repaired and checked by reading, compiled, not run on hardware
//
#include <hip/hip_runtime.h>

#define H 128
#define EFD 2
#define KIN 288      // 2H+EF=258 padded to multiple of 32
#define KN  256      // 2H
#define TE  32       // rows (edges or nodes) per tile
#define NLAYERS 2
#define INV_NORM (1.0f / 100.0f)
#define INT_OFF 106496   // LDS byte offset of inT in edge/coord kernels

typedef __attribute__((ext_vector_type(16))) __bf16 v16bf;
typedef __attribute__((ext_vector_type(8)))  __bf16 v8bf;
typedef __attribute__((ext_vector_type(8)))  float  v8f;

#if defined(__has_builtin)
#if __has_builtin(__builtin_amdgcn_tensor_load_to_lds) && __has_builtin(__builtin_amdgcn_s_wait_tensorcnt)
#define USE_TDM 1
#endif
#endif
#ifndef USE_TDM
#define USE_TDM 0
#endif

__device__ __forceinline__ float silu_f(float v) { return v / (1.0f + __expf(-v)); }
__device__ __forceinline__ float sigm_f(float v) { return 1.0f / (1.0f + __expf(-v)); }

// ---- WMMA fragment helpers (ISA 7.12.2 bf16 layouts) ----
__device__ __forceinline__ v16bf load_frag_A(const __bf16* base, int stride, int m0, int kb, int lane) {
  int m = m0 + (lane & 15);
  int half = lane >> 4;
  const __bf16* p = base + (size_t)m * stride + kb + half * 8;
  v8bf lo = *(const v8bf*)p;
  v8bf hi = *(const v8bf*)(p + 16);
  return __builtin_shufflevector(lo, hi, 0,1,2,3,4,5,6,7,8,9,10,11,12,13,14,15);
}

__device__ __forceinline__ v16bf load_frag_B(const __bf16* baseT, int stride, int n0, int kb, int lane) {
  int n = n0 + (lane & 15);
  int half = lane >> 4;
  const __bf16* p = baseT + (size_t)n * stride + kb + half * 8;
  v8bf lo = *(const v8bf*)p;
  v8bf hi = *(const v8bf*)(p + 16);
  return __builtin_shufflevector(lo, hi, 0,1,2,3,4,5,6,7,8,9,10,11,12,13,14,15);
}

__device__ __forceinline__ v8f wmma_mm(const __bf16* A, int As, int m0,
                                       const __bf16* Bt, int Bs, int n0,
                                       int K, int lane) {
  v8f acc = {};
  for (int kb = 0; kb < K; kb += 32) {
    v16bf a = load_frag_A(A, As, m0, kb, lane);
    v16bf b = load_frag_B(Bt, Bs, n0, kb, lane);
    acc = __builtin_amdgcn_wmma_f32_16x16x32_bf16(false, a, false, b, (short)0, acc, false, false);
  }
  return acc;
}

#if USE_TDM
typedef __attribute__((ext_vector_type(4))) unsigned int u32x4;
typedef __attribute__((ext_vector_type(4))) int i32x4;
typedef __attribute__((ext_vector_type(8))) int i32x8;

// TDM gather: 16 rows of 128 bf16 from hbf, scattered into LDS with 576B row
// stride (pad: 80 DWORDs after every 256B). 16-bit row indices (N < 65536).
__device__ __forceinline__ void tdm_gather16(const __bf16* hbf, unsigned lds_off,
                                             const int* idxp, int cnt, int N_) {
  int id[16];
#pragma unroll
  for (int i = 0; i < 16; ++i) id[i] = (i < cnt) ? idxp[i] : 0;
  i32x4 g2, g3;
#pragma unroll
  for (int j = 0; j < 4; ++j) g2[j] = (id[2 * j] & 0xFFFF) | (id[2 * j + 1] << 16);
#pragma unroll
  for (int j = 0; j < 4; ++j) g3[j] = (id[8 + 2 * j] & 0xFFFF) | (id[8 + 2 * j + 1] << 16);
  unsigned long long ga = (unsigned long long)(uintptr_t)hbf;
  u32x4 g0;
  g0[0] = 1u | (1u << 31);                        // count=1 | gather_mode (16b idx)
  g0[1] = lds_off;                                // lds_addr
  g0[2] = (unsigned)(ga & 0xFFFFFFFFu);           // global_addr lo
  g0[3] = (unsigned)(ga >> 32) | (2u << 30);      // global_addr hi | type=2
  i32x8 g1;
  g1[0] = (1 << 16) | (1 << 20) | (5 << 22) | (79 << 25); // data_size=2B, pad_en, 256B interval, 80DW pad
  g1[1] = (H & 0xFFFF) << 16;                     // tensor_dim0 lo (bits 63:48)
  g1[2] = (N_ & 0xFFFF) << 16;                    // tensor_dim0 hi | tensor_dim1 lo
  g1[3] = (N_ >> 16) | (H << 16);                 // tensor_dim1 hi | tile_dim0=128
  g1[4] = 16;                                     // tile_dim1 = #indices
  g1[5] = H;                                      // tensor_dim0_stride
  g1[6] = 0;
  g1[7] = 0;
  i32x8 gx = {};                                  // unused extended group (6-arg form)
  __builtin_amdgcn_tensor_load_to_lds(g0, g1, g2, g3, gx, 0);
}
#endif

// ---------------------------------------------------------------- prep
__global__ void prep_kernel(const float* __restrict__ x, const int* __restrict__ eidx,
                            const float* __restrict__ eattr,
                            float* __restrict__ cdiff, float* __restrict__ dist, int E_) {
  int e = blockIdx.x * 256 + threadIdx.x;
  if (e >= E_) return;
  int r = eidx[e], c = eidx[E_ + e];
  float dx = x[r * 3 + 0] - x[c * 3 + 0];
  float dy = x[r * 3 + 1] - x[c * 3 + 1];
  float dz = x[r * 3 + 2] - x[c * 3 + 2];
  float rad = dx * dx + dy * dy + dz * dz;
  float inv = 1.0f / (sqrtf(rad + 1e-8f) + 1.0f);
  cdiff[e * 3 + 0] = dx * inv;
  cdiff[e * 3 + 1] = dy * inv;
  cdiff[e * 3 + 2] = dz * inv;
  dist[e * 2 + 0] = rad;
  dist[e * 2 + 1] = eattr[e];
}

__global__ void h2bf_kernel(const float* __restrict__ h, __bf16* __restrict__ hbf, int n) {
  int i = blockIdx.x * 256 + threadIdx.x;
  if (i < n) hbf[i] = (__bf16)h[i];
}

// ---- shared gather of the edge-input tile (h[row] | h[col] | dist | pad) ----
__device__ __forceinline__ void gather_edge_tile(__bf16* inT, const __bf16* hbf,
                                                 const int* eidx, const float* dist,
                                                 int e0, int E_, int N_,
                                                 int tid, int wave) {
#if USE_TDM
  if (wave == 0) {
    for (int s = 0; s < TE / 16; ++s) {
      int base = e0 + s * 16;
      int cnt = E_ - base; cnt = cnt > 16 ? 16 : (cnt < 0 ? 0 : cnt);
      unsigned lds0 = (unsigned)(INT_OFF + s * 16 * (KIN * 2));
      tdm_gather16(hbf, lds0,         eidx + base,      cnt, N_);   // h[row] -> k 0..127
      tdm_gather16(hbf, lds0 + H * 2, eidx + E_ + base, cnt, N_);   // h[col] -> k 128..255
    }
    __builtin_amdgcn_s_wait_tensorcnt(0);
  }
#else
  for (int idx = tid; idx < TE * (2 * H) / 8; idx += 256) {  // 16B chunks
    int le = idx >> 5;
    int k0 = (idx & 31) * 8;
    int e = e0 + le;
    v8bf val = {};
    if (e < E_) {
      int node = (k0 < H) ? eidx[e] : eidx[E_ + e];
      val = *(const v8bf*)(hbf + (size_t)node * H + (k0 & (H - 1)));
    }
    *(v8bf*)(inT + le * KIN + k0) = val;
  }
#endif
  for (int idx = tid; idx < TE * 32; idx += 256) {           // dist cols + zero pad
    int le = idx >> 5, kk = idx & 31;
    int e = e0 + le;
    float v = 0.0f;
    if (e < E_ && kk < EFD) v = dist[(size_t)e * 2 + kk];
    inT[le * KIN + 2 * H + kk] = (__bf16)v;
  }
}

// ------------------------------------------------------------ edge MLP
// LDS: Wt1[128][288] | Wt2[128][128] | inT[32][288] | mid[32][128] (bf16)
//      mij[32][128] f32 | att[32] f32   (149632 B)
__global__ void __launch_bounds__(256)
edge_kernel(const __bf16* __restrict__ hbf, const int* __restrict__ eidx,
            const float* __restrict__ dist,
            const float* __restrict__ W1, const float* __restrict__ b1,
            const float* __restrict__ W2, const float* __restrict__ b2,
            const float* __restrict__ aw, const float* __restrict__ ab,
            float* __restrict__ agg, int E_, int N_) {
  extern __shared__ char smem[];
  __bf16* Wt1 = (__bf16*)(smem);
  __bf16* Wt2 = (__bf16*)(smem + 73728);
  __bf16* inT = (__bf16*)(smem + INT_OFF);
  __bf16* mid = (__bf16*)(smem + 124928);
  float*  mij = (float*)(smem + 133120);
  float*  att = (float*)(smem + 149504);

  int tid = threadIdx.x, lane = tid & 31, wave = tid >> 5;

  for (int idx = tid; idx < H * KIN; idx += 256) {
    int n = idx / KIN, k = idx - n * KIN;
    float v = (k < 2 * H + EFD) ? W1[k * H + n] : 0.0f;
    Wt1[idx] = (__bf16)v;
  }
  for (int idx = tid; idx < H * H; idx += 256) {
    int n = idx >> 7, k = idx & (H - 1);
    Wt2[idx] = (__bf16)W2[k * H + n];
  }
  __syncthreads();

  int ntiles = (E_ + TE - 1) / TE;
  for (int tile = blockIdx.x; tile < ntiles; tile += gridDim.x) {
    int e0 = tile * TE;
    gather_edge_tile(inT, hbf, eidx, dist, e0, E_, N_, tid, wave);
    __syncthreads();
    for (int t = wave; t < (TE / 16) * (H / 16); t += 8) {   // GEMM1 + silu
      int mt = t >> 3, nt = t & 7;
      v8f acc = wmma_mm(inT, KIN, mt * 16, Wt1, KIN, nt * 16, KIN, lane);
      int n = nt * 16 + (lane & 15);
      float bias = b1[n];
      int half = lane >> 4;
#pragma unroll
      for (int j = 0; j < 8; ++j) {
        int m = mt * 16 + j + 8 * half;
        mid[m * H + n] = (__bf16)silu_f(acc[j] + bias);
      }
    }
    __syncthreads();
    for (int t = wave; t < (TE / 16) * (H / 16); t += 8) {   // GEMM2 + silu
      int mt = t >> 3, nt = t & 7;
      v8f acc = wmma_mm(mid, H, mt * 16, Wt2, H, nt * 16, H, lane);
      int n = nt * 16 + (lane & 15);
      float bias = b2[n];
      int half = lane >> 4;
#pragma unroll
      for (int j = 0; j < 8; ++j) {
        int m = mt * 16 + j + 8 * half;
        mij[m * H + n] = silu_f(acc[j] + bias);
      }
    }
    __syncthreads();
    if (tid < TE) {                                          // attention
      float s = ab[0];
#pragma unroll 8
      for (int k = 0; k < H; ++k) s += mij[tid * H + k] * aw[k];
      att[tid] = sigm_f(s);
    }
    __syncthreads();
    for (int idx = tid; idx < TE * H; idx += 256) {          // scatter (1/NORM folded)
      int le = idx >> 7, n = idx & (H - 1);
      int e = e0 + le;
      if (e < E_) {
        float v = mij[idx] * att[le] * INV_NORM;
        atomicAdd(&agg[(size_t)eidx[e] * H + n], v);
      }
    }
    __syncthreads();
  }
}

// ------------------------------------------------------------ node MLP (residual, in-place h)
// LDS: Wt1[128][256] | Wt2[128][128] | inT[32][256] | mid[32][128]  (122880 B)
__global__ void __launch_bounds__(256)
node_kernel(float* __restrict__ hbuf, __bf16* __restrict__ hbf,
            const float* __restrict__ agg,
            const float* __restrict__ W1, const float* __restrict__ b1,
            const float* __restrict__ W2, const float* __restrict__ b2, int N_) {
  extern __shared__ char smem[];
  __bf16* Wt1 = (__bf16*)(smem);
  __bf16* Wt2 = (__bf16*)(smem + 65536);
  __bf16* inT = (__bf16*)(smem + 98304);
  __bf16* mid = (__bf16*)(smem + 114688);

  int tid = threadIdx.x, lane = tid & 31, wave = tid >> 5;

  for (int idx = tid; idx < H * KN; idx += 256) {
    int n = idx / KN, k = idx - n * KN;
    Wt1[idx] = (__bf16)W1[k * H + n];
  }
  for (int idx = tid; idx < H * H; idx += 256) {
    int n = idx >> 7, k = idx & (H - 1);
    Wt2[idx] = (__bf16)W2[k * H + n];
  }
  __syncthreads();

  int ntiles = (N_ + TE - 1) / TE;
  for (int tile = blockIdx.x; tile < ntiles; tile += gridDim.x) {
    int n0 = tile * TE;
    for (int idx = tid; idx < TE * H / 8; idx += 256) {      // h (bf16 mirror, 16B chunks)
      int ln = idx >> 4;
      int k0 = (idx & 15) * 8;
      int node = n0 + ln;
      v8bf val = {};
      if (node < N_) val = *(const v8bf*)(hbf + (size_t)node * H + k0);
      *(v8bf*)(inT + ln * KN + k0) = val;
    }
    for (int idx = tid; idx < TE * H; idx += 256) {          // agg (f32 -> bf16)
      int ln = idx >> 7, k = idx & (H - 1);
      int node = n0 + ln;
      float v = (node < N_) ? agg[(size_t)node * H + k] : 0.0f;
      inT[ln * KN + H + k] = (__bf16)v;
    }
    __syncthreads();
    for (int t = wave; t < (TE / 16) * (H / 16); t += 8) {   // GEMM1 + silu
      int mt = t >> 3, nt = t & 7;
      v8f acc = wmma_mm(inT, KN, mt * 16, Wt1, KN, nt * 16, KN, lane);
      int n = nt * 16 + (lane & 15);
      float bias = b1[n];
      int half = lane >> 4;
#pragma unroll
      for (int j = 0; j < 8; ++j) {
        int m = mt * 16 + j + 8 * half;
        mid[m * H + n] = (__bf16)silu_f(acc[j] + bias);
      }
    }
    __syncthreads();
    for (int t = wave; t < (TE / 16) * (H / 16); t += 8) {   // GEMM2 + bias + residual
      int mt = t >> 3, nt = t & 7;
      v8f acc = wmma_mm(mid, H, mt * 16, Wt2, H, nt * 16, H, lane);
      int n = nt * 16 + (lane & 15);
      float bias = b2[n];
      int half = lane >> 4;
#pragma unroll
      for (int j = 0; j < 8; ++j) {
        int m = mt * 16 + j + 8 * half;
        int node = n0 + m;
        if (node < N_) {
          float nv = hbuf[(size_t)node * H + n] + acc[j] + bias;  // unique (node,n)
          hbuf[(size_t)node * H + n] = nv;
          hbf[(size_t)node * H + n] = (__bf16)nv;                 // keep bf16 mirror fresh
        }
      }
    }
    __syncthreads();
  }
}

// ------------------------------------------------------------ coord MLP
__global__ void __launch_bounds__(256)
coord_kernel(const __bf16* __restrict__ hbf, const int* __restrict__ eidx,
             const float* __restrict__ dist, const float* __restrict__ cdiff,
             const float* __restrict__ W1, const float* __restrict__ b1,
             const float* __restrict__ W2, const float* __restrict__ b2,
             const float* __restrict__ W3,
             float* __restrict__ xout, int E_, int N_) {
  extern __shared__ char smem[];
  __bf16* Wt1 = (__bf16*)(smem);
  __bf16* Wt2 = (__bf16*)(smem + 73728);
  __bf16* inT = (__bf16*)(smem + INT_OFF);
  __bf16* mid = (__bf16*)(smem + 124928);
  float*  mij = (float*)(smem + 133120);
  float*  phi = (float*)(smem + 149504);

  int tid = threadIdx.x, lane = tid & 31, wave = tid >> 5;

  for (int idx = tid; idx < H * KIN; idx += 256) {
    int n = idx / KIN, k = idx - n * KIN;
    float v = (k < 2 * H + EFD) ? W1[k * H + n] : 0.0f;
    Wt1[idx] = (__bf16)v;
  }
  for (int idx = tid; idx < H * H; idx += 256) {
    int n = idx >> 7, k = idx & (H - 1);
    Wt2[idx] = (__bf16)W2[k * H + n];
  }
  __syncthreads();

  int ntiles = (E_ + TE - 1) / TE;
  for (int tile = blockIdx.x; tile < ntiles; tile += gridDim.x) {
    int e0 = tile * TE;
    gather_edge_tile(inT, hbf, eidx, dist, e0, E_, N_, tid, wave);
    __syncthreads();
    for (int t = wave; t < (TE / 16) * (H / 16); t += 8) {
      int mt = t >> 3, nt = t & 7;
      v8f acc = wmma_mm(inT, KIN, mt * 16, Wt1, KIN, nt * 16, KIN, lane);
      int n = nt * 16 + (lane & 15);
      float bias = b1[n];
      int half = lane >> 4;
#pragma unroll
      for (int j = 0; j < 8; ++j) {
        int m = mt * 16 + j + 8 * half;
        mid[m * H + n] = (__bf16)silu_f(acc[j] + bias);
      }
    }
    __syncthreads();
    for (int t = wave; t < (TE / 16) * (H / 16); t += 8) {
      int mt = t >> 3, nt = t & 7;
      v8f acc = wmma_mm(mid, H, mt * 16, Wt2, H, nt * 16, H, lane);
      int n = nt * 16 + (lane & 15);
      float bias = b2[n];
      int half = lane >> 4;
#pragma unroll
      for (int j = 0; j < 8; ++j) {
        int m = mt * 16 + j + 8 * half;
        mij[m * H + n] = silu_f(acc[j] + bias);
      }
    }
    __syncthreads();
    if (tid < TE) {                               // phi = mij @ cw3 (no bias/act)
      float s = 0.0f;
#pragma unroll 8
      for (int k = 0; k < H; ++k) s += mij[tid * H + k] * W3[k];
      phi[tid] = s;
    }
    __syncthreads();
    for (int idx = tid; idx < TE * 3; idx += 256) {   // x += cdiff * phi / NORM
      int le = idx / 3, d = idx - le * 3;
      int e = e0 + le;
      if (e < E_) {
        float v = cdiff[(size_t)e * 3 + d] * phi[le] * INV_NORM;
        atomicAdd(&xout[(size_t)eidx[e] * 3 + d], v);
      }
    }
    __syncthreads();
  }
}

// ---------------------------------------------------------------- launch
extern "C" void kernel_launch(void* const* d_in, const int* in_sizes, int n_in,
                              void* d_out, int out_size, void* d_ws, size_t ws_size,
                              hipStream_t stream) {
  const float* h     = (const float*)d_in[0];
  const float* x     = (const float*)d_in[1];
  const int*   eidx  = (const int*)d_in[2];
  const float* eattr = (const float*)d_in[3];
  const float* ew1 = (const float*)d_in[4];  const float* eb1 = (const float*)d_in[5];
  const float* ew2 = (const float*)d_in[6];  const float* eb2 = (const float*)d_in[7];
  const float* nw1 = (const float*)d_in[8];  const float* nb1 = (const float*)d_in[9];
  const float* nw2 = (const float*)d_in[10]; const float* nb2 = (const float*)d_in[11];
  const float* aw  = (const float*)d_in[12]; const float* ab  = (const float*)d_in[13];
  const float* cw1 = (const float*)d_in[14]; const float* cb1 = (const float*)d_in[15];
  const float* cw2 = (const float*)d_in[16]; const float* cb2 = (const float*)d_in[17];
  const float* cw3 = (const float*)d_in[18];

  const int N = in_sizes[0] / H;
  const int E = in_sizes[3];

  float* out_h = (float*)d_out;                       // [N,H]
  float* out_x = (float*)d_out + (size_t)N * H;       // [N,3]

  char* ws = (char*)d_ws;
  float*  agg   = (float*)ws;                                           // N*H f32
  float*  dist  = (float*)(ws + (size_t)N * H * 4);                     // E*2 f32
  float*  cdiff = (float*)(ws + (size_t)N * H * 4 + (size_t)E * 2 * 4); // E*3 f32
  __bf16* hbf   = (__bf16*)(ws + (size_t)N * H * 4 + (size_t)E * 5 * 4);// N*H bf16

  (void)hipMemcpyAsync(out_h, h, (size_t)N * H * sizeof(float), hipMemcpyDeviceToDevice, stream);
  (void)hipMemcpyAsync(out_x, x, (size_t)N * 3 * sizeof(float), hipMemcpyDeviceToDevice, stream);

  prep_kernel<<<(E + 255) / 256, 256, 0, stream>>>(x, eidx, eattr, cdiff, dist, E);
  h2bf_kernel<<<(N * H + 255) / 256, 256, 0, stream>>>(h, hbf, N * H);

  const int etiles = (E + TE - 1) / TE;
  const int ntilesN = (N + TE - 1) / TE;
  const int egrid = etiles < 2048 ? etiles : 2048;
  const int ngrid = ntilesN < 1024 ? ntilesN : 1024;
  const size_t EDGE_SMEM = 149632, NODE_SMEM = 122880, COORD_SMEM = 149632;

  for (int l = 0; l < NLAYERS; ++l) {
    (void)hipMemsetAsync(agg, 0, (size_t)N * H * sizeof(float), stream);
    edge_kernel<<<egrid, 256, EDGE_SMEM, stream>>>(
        hbf, eidx, dist,
        ew1 + (size_t)l * (2 * H + EFD) * H, eb1 + (size_t)l * H,
        ew2 + (size_t)l * H * H,             eb2 + (size_t)l * H,
        aw  + (size_t)l * H,                 ab  + l,
        agg, E, N);
    node_kernel<<<ngrid, 256, NODE_SMEM, stream>>>(
        out_h, hbf, agg,
        nw1 + (size_t)l * 2 * H * H, nb1 + (size_t)l * H,
        nw2 + (size_t)l * H * H,     nb2 + (size_t)l * H, N);
  }

  coord_kernel<<<egrid, 256, COORD_SMEM, stream>>>(
      hbf, eidx, dist, cdiff, cw1, cb1, cw2, cb2, cw3, out_x, E, N);
}
